// RESCAL_69544110456887
// MI455X (gfx1250) — matrix operand order repaired and checked by reading, not compiled, not run on hardware
//
#include <hip/hip_runtime.h>

// RESCAL margin loss on MI455X (gfx1250, wave32, WMMA).
// One wave per (pos,neg) triple pair. Per triple: v = R t via
// v_wmma_f32_16x16x32_bf16 (4 row panels x 2 K-chunks), t replicated across
// the 16 B-columns; then score = h.v / 64 via per-lane FMAs + wave reduction.

#define DD 64

typedef __attribute__((ext_vector_type(16))) __bf16 v16bf;
typedef __attribute__((ext_vector_type(8)))  float  v8f;
typedef __attribute__((ext_vector_type(4)))  float  v4f;

__device__ __forceinline__ float triple_score(const float* __restrict__ ent,
                                              const float* __restrict__ rel,
                                              int h_idx, int t_idx, int r_idx,
                                              int lane)
{
    const float* __restrict__ hrow = ent + (size_t)h_idx * DD;
    const float* __restrict__ trow = ent + (size_t)t_idx * DD;
    const float* __restrict__ Rm   = rel + (size_t)r_idx * (DD * DD);

    const int m  = lane & 15;   // row within panel (A) / column (B,C)
    const int hi = lane >> 4;   // half-wave select per ISA VGPR layouts

    // B-matrix (32x16 bf16), two K-chunks, t replicated across all N columns.
    // k order used here: VGPRs 0..7, lanes 0-15 hold K=0..15, lanes 16-31 hold K=16..31.
    v16bf Bq[2];
#pragma unroll
    for (int q = 0; q < 2; ++q) {
        const float* tb = trow + 32 * q + hi * 16;
        v4f b0 = *(const v4f*)(tb + 0);
        v4f b1 = *(const v4f*)(tb + 4);
        v4f b2 = *(const v4f*)(tb + 8);
        v4f b3 = *(const v4f*)(tb + 12);
#pragma unroll
        for (int k = 0; k < 4; ++k) {
            Bq[q][k +  0] = (__bf16)b0[k];
            Bq[q][k +  4] = (__bf16)b1[k];
            Bq[q][k +  8] = (__bf16)b2[k];
            Bq[q][k + 12] = (__bf16)b3[k];
        }
    }

    float acc = 0.0f;
#pragma unroll
    for (int p = 0; p < 4; ++p) {         // 16-row panels of R
        v8f c = {};                        // f32 accumulator tile
#pragma unroll
        for (int q = 0; q < 2; ++q) {     // K chunks of 32
            // A-matrix 16x32 bf16, k order used here: lane m (hi=0) holds K 0..15;
            // lane m+16 (hi=1) holds K 16..31 of row M=m.
            const float* ar = Rm + (size_t)(16 * p + m) * DD + 32 * q + hi * 16;
            v4f a0 = *(const v4f*)(ar + 0);
            v4f a1 = *(const v4f*)(ar + 4);
            v4f a2 = *(const v4f*)(ar + 8);
            v4f a3 = *(const v4f*)(ar + 12);
            v16bf A;
#pragma unroll
            for (int k = 0; k < 4; ++k) {
                A[k +  0] = (__bf16)a0[k];
                A[k +  4] = (__bf16)a1[k];
                A[k +  8] = (__bf16)a2[k];
                A[k + 12] = (__bf16)a3[k];
            }
            // (neg_a, A, neg_b, B, c_mod, C, reuse_a, reuse_b)
            c = __builtin_amdgcn_wmma_f32_16x16x32_bf16(
                    false, A, false, Bq[q], (short)0, c, false, false);
        }
        // C layout: VGPR r holds row (16p + r + 8*hi); every column identical.
#pragma unroll
        for (int r2 = 0; r2 < 8; ++r2)
            acc += hrow[16 * p + r2 + 8 * hi] * c[r2];
    }

    // Wave32 reduction. Columns are 16x replicated -> divide by 16; mean over
    // D=64 -> divide by 64.
#pragma unroll
    for (int off = 16; off > 0; off >>= 1)
        acc += __shfl_xor(acc, off, 32);
    return acc * (1.0f / (16.0f * 64.0f));
}

__global__ void rescal_zero_out(float* out)
{
    if (threadIdx.x == 0) out[0] = 0.0f;
}

__global__ __launch_bounds__(256)
void rescal_margin_kernel(const float* __restrict__ ent,
                          const float* __restrict__ rel,
                          const int* __restrict__ ph, const int* __restrict__ pt,
                          const int* __restrict__ pr, const int* __restrict__ nh,
                          const int* __restrict__ nt, const int* __restrict__ nr,
                          float* __restrict__ out, int n)
{
    const int wid  = (blockIdx.x * blockDim.x + threadIdx.x) >> 5;  // wave id
    const int lane = threadIdx.x & 31;
    if (wid >= n) return;   // wave-uniform: EXEC all-1s at every WMMA

    const int iph = ph[wid], ipt = pt[wid], ipr = pr[wid];
    const int inh = nh[wid], int_ = nt[wid], inr = nr[wid];

    // Warm L2 for both relation matrices (16 KB each; 16 MB total set fits
    // easily in the 192 MB L2). Emits global_prefetch_b8.
    __builtin_prefetch(rel + (size_t)ipr * (DD * DD), 0, 3);
    __builtin_prefetch(rel + (size_t)inr * (DD * DD), 0, 3);

    float sp = triple_score(ent, rel, iph, ipt, ipr, lane);
    float sn = triple_score(ent, rel, inh, int_, inr, lane);

    float loss = fmaxf(0.0f, sn - sp + 1.0f);
    if (lane == 0) atomicAdd(out, loss);
}

extern "C" void kernel_launch(void* const* d_in, const int* in_sizes, int n_in,
                              void* d_out, int out_size, void* d_ws, size_t ws_size,
                              hipStream_t stream)
{
    const float* ent = (const float*)d_in[0];
    const float* rel = (const float*)d_in[1];
    const int*   ph  = (const int*)d_in[2];
    const int*   pt  = (const int*)d_in[3];
    const int*   pr  = (const int*)d_in[4];
    const int*   nh  = (const int*)d_in[5];
    const int*   nt  = (const int*)d_in[6];
    const int*   nr  = (const int*)d_in[7];
    float* out = (float*)d_out;

    const int n = in_sizes[2];                 // B triples
    rescal_zero_out<<<1, 32, 0, stream>>>(out);

    const int waves_per_block = 8;             // 256 threads = 8 wave32
    const int blocks = (n + waves_per_block - 1) / waves_per_block;
    rescal_margin_kernel<<<blocks, 256, 0, stream>>>(ent, rel, ph, pt, pr,
                                                     nh, nt, nr, out, n);
}
